// DecoderLSTM_75076028334172
// MI455X (gfx1250) — matrix operand order repaired and checked by reading, not compiled, not run on hardware
//
#include <hip/hip_runtime.h>

// Problem sizes (fixed by the reference)
#define TSB 32     // batch B
#define TT  128    // T
#define TE  512    // E
#define THID 1024  // H
#define TV  32000  // V
#define TSTEPS 127 // T-1
#define NGATE 4096 // 4H

typedef _Float16 v8h  __attribute__((ext_vector_type(8)));
typedef _Float16 v16h __attribute__((ext_vector_type(16)));
typedef float    v8f  __attribute__((ext_vector_type(8)));

union AFrag { v16h v; v8h h[2]; };

// One pipeline stage: 32-row A strip + two 16-col B tiles (32 VGPRs total)
struct F2 {
  AFrag a0, a1;
  v16h b0, b1;
};

__device__ __forceinline__ float sigmoidf_(float x) {
  return 1.0f / (1.0f + __expf(-x));
}

__device__ __forceinline__ void load_f2(F2& f,
    const _Float16* a0p, const _Float16* a1p,
    const _Float16* b0p, const _Float16* b1p) {
  f.a0.h[0] = *(const v8h*)(a0p);
  f.a0.h[1] = *(const v8h*)(a0p + 8);
  f.a1.h[0] = *(const v8h*)(a1p);
  f.a1.h[1] = *(const v8h*)(a1p + 8);
  f.b0 = *(const v16h*)(b0p);
  f.b1 = *(const v16h*)(b1p);
}

__device__ __forceinline__ void wmma_f2(const F2& f, v8f acc[2][2]) {
  acc[0][0] = __builtin_amdgcn_wmma_f32_16x16x32_f16(false, f.a0.v, false, f.b0, (short)0, acc[0][0], false, false);
  acc[1][0] = __builtin_amdgcn_wmma_f32_16x16x32_f16(false, f.a1.v, false, f.b0, (short)0, acc[1][0], false, false);
  acc[0][1] = __builtin_amdgcn_wmma_f32_16x16x32_f16(false, f.a0.v, false, f.b1, (short)0, acc[0][1], false, false);
  acc[1][1] = __builtin_amdgcn_wmma_f32_16x16x32_f16(false, f.a1.v, false, f.b1, (short)0, acc[1][1], false, false);
}

// acc[mh][j] += A(32 x Klen slice, row stride ld) @ W(rows row0/row1 + nn, slice)^T
// One-deep software pipeline; Klen must be a multiple of 64.
// Fragment layouts per CDNA5 ISA 7.12.2 (16-bit A 16x32; B 32x16; C/D 16x16 f32).
__device__ __forceinline__ void mm_acc2(const _Float16* __restrict__ A,
                                        const _Float16* __restrict__ W,
                                        int ld, int Klen,
                                        int row0, int row1,
                                        int lane, v8f acc[2][2]) {
  const int m   = lane & 15;
  const int akb = (lane < 16) ? 0 : 16;  // A: K sub-chunk base per lane half
  const int nn  = lane & 15;
  const int bkb = (lane < 16) ? 0 : 16;  // B: K base per lane half
  const _Float16* a0p = A + (size_t)m * ld + akb;
  const _Float16* a1p = A + (size_t)(m + 16) * ld + akb;
  const _Float16* b0p = W + (size_t)(row0 + nn) * ld + bkb;
  const _Float16* b1p = W + (size_t)(row1 + nn) * ld + bkb;

  F2 f0, f1;
  load_f2(f0, a0p, a1p, b0p, b1p);
  a0p += 32; a1p += 32; b0p += 32; b1p += 32;

  for (int kt = 0; kt < Klen; kt += 64) {
    load_f2(f1, a0p, a1p, b0p, b1p);       // prefetch kt+32
    a0p += 32; a1p += 32; b0p += 32; b1p += 32;
    wmma_f2(f0, acc);
    if (kt + 64 < Klen) {
      load_f2(f0, a0p, a1p, b0p, b1p);     // prefetch kt+64
      a0p += 32; a1p += 32; b0p += 32; b1p += 32;
    }
    wmma_f2(f1, acc);
  }
}

// --- small prep kernels -----------------------------------------------------

__global__ __launch_bounds__(256) void embed_kernel(
    const int* __restrict__ cap, const float* __restrict__ table,
    const float* __restrict__ pos, _Float16* __restrict__ Xh) {
  size_t i = (size_t)blockIdx.x * blockDim.x + threadIdx.x;
  if (i >= (size_t)TSTEPS * TSB * TE) return;
  int e = (int)(i % TE);
  int r = (int)(i / TE);
  int b = r % TSB;
  int t = r / TSB;
  int tok = cap[b * TT + t];                        // captions[:, t], t in [0, T-2]
  Xh[i] = (_Float16)(table[(size_t)tok * TE + e] + pos[e]);
}

__global__ __launch_bounds__(256) void f32_to_f16_kernel(
    const float* __restrict__ src, _Float16* __restrict__ dst, int n) {
  int i = blockIdx.x * blockDim.x + threadIdx.x;
  if (i < n) dst[i] = (_Float16)src[i];
}

__global__ __launch_bounds__(256) void zero_t0_kernel(float* __restrict__ out) {
  int i = blockIdx.x * blockDim.x + threadIdx.x;
  if (i >= TSB * TV) return;
  int b = i / TV, v = i % TV;
  out[(size_t)b * TT * TV + v] = 0.0f;              // outputs[:,0,:] = 0
}

// --- one LSTM layer, one timestep ------------------------------------------
// 64 blocks (one per 16-unit hidden strip) x 4 waves. Wave (p, kh):
//   p  = gate pair (0 -> i,f ; 1 -> g,o), kh = K half.
// Partial gate sums meet in LDS; cell update is fragment-layout free.
__global__ __launch_bounds__(128) void lstm_step_kernel(
    const _Float16* __restrict__ X, int Kx,
    const _Float16* __restrict__ Hprev,
    const _Float16* __restrict__ Wih,
    const _Float16* __restrict__ Whh,
    const float* __restrict__ b_ih,
    const float* __restrict__ b_hh,
    float* __restrict__ cst,
    _Float16* __restrict__ hout,
    _Float16* __restrict__ hsout) {
  __shared__ float gbuf[2][4][TSB][16];   // [kh][gate][batch][unit] = 16 KB
  const int hu   = blockIdx.x;            // 0..63
  const int wid  = threadIdx.x >> 5;      // 0..3
  const int lane = threadIdx.x & 31;
  const int p    = wid & 1;               // gate pair
  const int kh   = wid >> 1;              // K half

  const int row0 = (p * 2 + 0) * THID + hu * 16;  // W rows for this wave's gates
  const int row1 = (p * 2 + 1) * THID + hu * 16;

  v8f acc[2][2] = {};
  // x_t @ W_ih^T : this wave reduces K slice [kh*Kx/2, (kh+1)*Kx/2)
  mm_acc2(X + kh * (Kx >> 1), Wih + kh * (Kx >> 1), Kx, Kx >> 1, row0, row1, lane, acc);
  // h_{t-1} @ W_hh^T : K slice of H
  mm_acc2(Hprev + kh * (THID >> 1), Whh + kh * (THID >> 1), THID, THID >> 1, row0, row1, lane, acc);

  // scatter partial sums to LDS per D-fragment layout
  const int nn = lane & 15;
  const int mbase = (lane < 16) ? 0 : 8;
#pragma unroll
  for (int g = 0; g < 2; ++g) {
#pragma unroll
    for (int mh = 0; mh < 2; ++mh) {
#pragma unroll
      for (int r = 0; r < 8; ++r) {
        const int b = mh * 16 + mbase + r;
        gbuf[kh][p * 2 + g][b][nn] = acc[mh][g][r];
      }
    }
  }
  __syncthreads();

  // cell update: 512 cells, 4 per thread (lane-stride-1 -> conflict-free LDS)
#pragma unroll
  for (int j = 0; j < 4; ++j) {
    const int cell = threadIdx.x + j * 128;   // 0..511
    const int b  = cell >> 4;
    const int ul = cell & 15;
    const int u  = hu * 16 + ul;
    const float gi = gbuf[0][0][b][ul] + gbuf[1][0][b][ul] + b_ih[u] + b_hh[u];
    const float gf = gbuf[0][1][b][ul] + gbuf[1][1][b][ul] + b_ih[THID + u] + b_hh[THID + u];
    const float gg = gbuf[0][2][b][ul] + gbuf[1][2][b][ul] + b_ih[2 * THID + u] + b_hh[2 * THID + u];
    const float go = gbuf[0][3][b][ul] + gbuf[1][3][b][ul] + b_ih[3 * THID + u] + b_hh[3 * THID + u];
    const size_t idx = (size_t)b * THID + u;
    const float cnew = sigmoidf_(gf) * cst[idx] + sigmoidf_(gi) * tanhf(gg);
    cst[idx] = cnew;
    const _Float16 hval = (_Float16)(sigmoidf_(go) * tanhf(cnew));
    hout[idx] = hval;
    if (hsout) hsout[idx] = hval;
  }
}

// --- output projection: (127*32, 1024) @ (32000, 1024)^T -> d_out -----------
// Each wave computes a 32(M) x 32(N) tile. Waves within a block take the SAME
// vocab tile (nt) and consecutive timesteps (mt), so B-weight loads hit WGP$.
__global__ __launch_bounds__(256) void out_proj_kernel(
    const _Float16* __restrict__ HS,
    const _Float16* __restrict__ Wo,
    const float* __restrict__ bo,
    float* __restrict__ out) {
  const int wave = (blockIdx.x * blockDim.x + threadIdx.x) >> 5;
  const int lane = threadIdx.x & 31;
  const int NT = TV / 32;           // 1000 column tiles
  if (wave >= TSTEPS * NT) return;  // wave-uniform exit (safe with WMMA)
  const int nt = wave / TSTEPS;     // vocab tile (shared within a block)
  const int mt = wave % TSTEPS;     // timestep / 32-row strip

  v8f acc[2][2] = {};
  mm_acc2(HS + (size_t)mt * TSB * THID, Wo, THID, THID,
          nt * 32, nt * 32 + 16, lane, acc);

  const int nn = lane & 15;
  const int mbase = (lane < 16) ? 0 : 8;
#pragma unroll
  for (int g = 0; g < 2; ++g) {
    const int v = nt * 32 + g * 16 + nn;
    const float bias = bo[v];
#pragma unroll
    for (int mh = 0; mh < 2; ++mh) {
#pragma unroll
      for (int r = 0; r < 8; ++r) {
        const int b = mh * 16 + mbase + r;            // batch row
        out[((size_t)b * TT + (mt + 1)) * TV + v] = acc[mh][g][r] + bias;
      }
    }
  }
}

// ---------------------------------------------------------------------------

extern "C" void kernel_launch(void* const* d_in, const int* in_sizes, int n_in,
                              void* d_out, int out_size, void* d_ws, size_t ws_size,
                              hipStream_t stream) {
  (void)in_sizes; (void)n_in; (void)out_size; (void)ws_size;

  const int*   captions = (const int*)  d_in[1];
  const float* table    = (const float*)d_in[2];
  const float* pos      = (const float*)d_in[3];
  const float* W_ih0    = (const float*)d_in[4];
  const float* W_hh0    = (const float*)d_in[5];
  const float* b_ih0    = (const float*)d_in[6];
  const float* b_hh0    = (const float*)d_in[7];
  const float* W_ih1    = (const float*)d_in[8];
  const float* W_hh1    = (const float*)d_in[9];
  const float* b_ih1    = (const float*)d_in[10];
  const float* b_hh1    = (const float*)d_in[11];
  const float* W_out    = (const float*)d_in[12];
  const float* b_out    = (const float*)d_in[13];
  float* out = (float*)d_out;

  // Workspace carve-out (256B aligned slabs)
  char* ws = (char*)d_ws;
  size_t off = 0;
  auto take = [&](size_t bytes) -> void* {
    void* p = ws + off;
    off += (bytes + 255) & ~(size_t)255;
    return p;
  };
  _Float16* Xh     = (_Float16*)take((size_t)TSTEPS * TSB * TE * 2);
  _Float16* Wih0h  = (_Float16*)take((size_t)NGATE * TE * 2);
  _Float16* Whh0h  = (_Float16*)take((size_t)NGATE * THID * 2);
  _Float16* Wih1h  = (_Float16*)take((size_t)NGATE * THID * 2);
  _Float16* Whh1h  = (_Float16*)take((size_t)NGATE * THID * 2);
  _Float16* Wouth  = (_Float16*)take((size_t)TV * THID * 2);
  // state region (zeroed each call with one memset): h0 ping-pong, h1 ping-pong, c0, c1
  char* state_base = (char*)take(0);
  _Float16* h0buf[2];
  h0buf[0] = (_Float16*)take((size_t)TSB * THID * 2);
  h0buf[1] = (_Float16*)take((size_t)TSB * THID * 2);
  _Float16* h1buf[2];
  h1buf[0] = (_Float16*)take((size_t)TSB * THID * 2);
  h1buf[1] = (_Float16*)take((size_t)TSB * THID * 2);
  float* c0 = (float*)take((size_t)TSB * THID * 4);
  float* c1 = (float*)take((size_t)TSB * THID * 4);
  size_t state_bytes = (size_t)((char*)take(0) - state_base);
  _Float16* HS = (_Float16*)take((size_t)TSTEPS * TSB * THID * 2);

  // 1) zero recurrent state (must happen every call: ws is not re-poisoned)
  hipMemsetAsync(state_base, 0, state_bytes, stream);

  // 2) embeddings + pos_enc -> f16, time-major
  {
    size_t n = (size_t)TSTEPS * TSB * TE;
    embed_kernel<<<(unsigned)((n + 255) / 256), 256, 0, stream>>>(captions, table, pos, Xh);
  }

  // 3) weights fp32 -> f16
  {
    struct { const float* s; _Float16* d; int n; } cv[5] = {
      { W_ih0, Wih0h, NGATE * TE  },
      { W_hh0, Whh0h, NGATE * THID },
      { W_ih1, Wih1h, NGATE * THID },
      { W_hh1, Whh1h, NGATE * THID },
      { W_out, Wouth, TV * THID   },
    };
    for (int i = 0; i < 5; ++i)
      f32_to_f16_kernel<<<(cv[i].n + 255) / 256, 256, 0, stream>>>(cv[i].s, cv[i].d, cv[i].n);
  }

  // 4) sequential 2-layer LSTM scan (two kernels per step; stream order = dependency)
  for (int t = 0; t < TSTEPS; ++t) {
    const _Float16* h0p = h0buf[t & 1];
    _Float16*       h0n = h0buf[(t + 1) & 1];
    const _Float16* h1p = h1buf[t & 1];
    _Float16*       h1n = h1buf[(t + 1) & 1];
    lstm_step_kernel<<<64, 128, 0, stream>>>(
        Xh + (size_t)t * TSB * TE, TE, h0p, Wih0h, Whh0h, b_ih0, b_hh0,
        c0, h0n, (_Float16*)nullptr);
    lstm_step_kernel<<<64, 128, 0, stream>>>(
        h0n, THID, h1p, Wih1h, Whh1h, b_ih1, b_hh1,
        c1, h1n, HS + (size_t)t * TSB * THID);
  }

  // 5) vocab projection (dominant GEMM) + zero the t=0 slot
  {
    const int NT = TV / 32;
    int waves = TSTEPS * NT;              // 127000
    int blocks = (waves + 7) / 8;         // 8 waves / 256-thread block
    out_proj_kernel<<<blocks, 256, 0, stream>>>(HS, Wouth, b_out, out);
    zero_t0_kernel<<<(TSB * TV + 255) / 256, 256, 0, stream>>>(out);
  }
}
